// DiffeqSolver_15685220565687
// MI455X (gfx1250) — matrix operand order repaired and checked by reading, not compiled, not run on hardware
//
#include <hip/hip_runtime.h>
#include <math.h>

// ---------------------------------------------------------------------------
// SPDE spectral Euler solver on gfx1250 (MI455X), all GEMMs via
// v_wmma_f32_16x16x32_bf16 (bf16 operands, f32 accumulate).
// Shapes: BATCH=16, HIDDEN=128, DIM_X=128, N_FORCING=4, DIM_T=64, MODES=96.
// fftshift/ifftshift/band-limit folded into 96-mode DFT matrices.
// ---------------------------------------------------------------------------

#define BATCH  16
#define HID    128
#define DX     128
#define NF     4
#define DT     64
#define MODES  96
#define NSTEP  63           // DT-1 Euler steps
#define ROWS   (BATCH*HID)  // 2048 combined (b,h) rows

typedef __bf16 bf16_t;
typedef __attribute__((ext_vector_type(16))) __bf16 v16bf;
typedef __attribute__((ext_vector_type(8)))  __bf16 v8bf;
typedef __attribute__((ext_vector_type(8)))  float  v8f;
typedef __attribute__((ext_vector_type(4)))  float  v4f;

static __device__ __forceinline__ v8bf ld8(const bf16_t* p) {
    return *reinterpret_cast<const v8bf*>(p);
}
static __device__ __forceinline__ v16bf ld16(const bf16_t* p) {
    return *reinterpret_cast<const v16bf*>(p);
}
// WMMA A-operand (16x32 bf16): lane<16 holds K {k0..k0+15},
// lane>=16 holds K {k0+16..k0+31} (matches B). Two b128 loads.
static __device__ __forceinline__ v16bf joinA(const bf16_t* rowp, int k0, int ksel) {
    v8bf lo = ld8(rowp + k0 + 2 * ksel);
    v8bf hi = ld8(rowp + k0 + 2 * ksel + 8);
    v16bf r;
#pragma unroll
    for (int i = 0; i < 8; ++i) { r[i] = lo[i]; r[i + 8] = hi[i]; }
    return r;
}
static __device__ __forceinline__ v8f wmma_bf16(v16bf a, v16bf b, v8f c) {
    return __builtin_amdgcn_wmma_f32_16x16x32_bf16(false, a, false, b, (short)0, c,
                                                   false, false);
}
static __device__ __forceinline__ void store8(float* p, v8f v) {
    v4f a = {v[0], v[1], v[2], v[3]};
    v4f b = {v[4], v[5], v[6], v[7]};
    *reinterpret_cast<v4f*>(p)     = a;
    *reinterpret_cast<v4f*>(p + 4) = b;
}

// ------------------------------- init kernels ------------------------------

__global__ void k_cvt(const float* __restrict__ src, bf16_t* __restrict__ dst, int n) {
    int i = blockIdx.x * blockDim.x + threadIdx.x;
    if (i < n) dst[i] = (bf16_t)src[i];
}

// Wcat rows 0..127 = W_F[o][h]; rows 128..639 = W_G[o][n][h] flat (o*4+n).
__global__ void k_wcat(const float* __restrict__ WF, const float* __restrict__ WG,
                       bf16_t* __restrict__ Wcat) {
    int i = blockIdx.x * blockDim.x + threadIdx.x;
    if (i >= 640 * HID) return;
    int r = i / HID, h = i % HID;
    float v = (r < HID) ? WF[r * HID + h] : WG[(r - HID) * HID + h];
    Wcat[i] = (bf16_t)v;
}

// Band DFT twiddles. Shifted band position m in [0,96) <-> FFT bin km=(m+80)%128.
// Forward:  hf[m] = sum_x H[x] * (cos - i sin),  theta = 2*pi*km*x/128
// Inverse:  z[x]  = (1/128) * sum_m (vre*cos - vim*sin)
__global__ void k_twiddle(bf16_t* __restrict__ Wfre, bf16_t* __restrict__ Wfim,
                          bf16_t* __restrict__ Btre, bf16_t* __restrict__ Btim) {
    int idx = blockIdx.x * blockDim.x + threadIdx.x;
    if (idx >= DX * MODES) return;
    int m = idx % MODES, x = idx / MODES;
    int km = (m + 80) & 127;
    int ph = (km * x) & 127;                         // exact phase mod 128
    float th = (float)ph * (6.2831853071795864769f / 128.0f);
    float c = cosf(th), s = sinf(th);
    Wfre[m * DX + x]   = (bf16_t)c;                  // forward, K=x contiguous
    Wfim[m * DX + x]   = (bf16_t)(-s);
    Btre[x * MODES + m] = (bf16_t)(c * (1.0f / 128.0f)); // inverse, K=m contiguous
    Btim[x * MODES + m] = (bf16_t)(-s * (1.0f / 128.0f));
}

// dxi[t][b][x][n] = xi[b][n][x][t+1] - xi[b][n][x][t]
__global__ void k_dxi(const float* __restrict__ xi, float* __restrict__ dxi) {
    int i = blockIdx.x * blockDim.x + threadIdx.x;
    if (i >= NSTEP * BATCH * DX * NF) return;
    int n = i & 3, x = (i >> 2) & 127, b = (i >> 9) & 15, t = i >> 13;
    const float* p = xi + (((size_t)(b * NF + n) * DX + x) * DT + t);
    dxi[i] = p[1] - p[0];
}

// State v: f32 layout [m][c][b*128+h]. Produce bf16 copies in two layouts:
// vbh (same order, K=h contiguous for the mode matvec B-operand) and
// vbm ([c][row][m], K=m contiguous for the IDFT A-operand).
__global__ void k_cvtv(const float* __restrict__ v, bf16_t* __restrict__ vbh,
                       bf16_t* __restrict__ vbm) {
    int i = blockIdx.x * blockDim.x + threadIdx.x;
    if (i >= MODES * 2 * ROWS) return;
    float f = v[i];
    vbh[i] = (bf16_t)f;
    int row = i & 2047, c = (i >> 11) & 1, m = i >> 12;
    vbm[((size_t)c * ROWS + row) * MODES + m] = (bf16_t)f;
}

// ------------------------- per-mode complex matvec -------------------------
// Av_re = A0@v_re - A1@v_im ; Av_im = A1@v_re - A0@v_im  (reference's signs).
// Per mode: M=128 (h_out), N=16 (batch), K=128 (h_in). Grid: 96 blocks x 8 waves,
// wave = M-tile; 4 accumulators, 16 WMMAs per wave.
__global__ void __launch_bounds__(256) k_modes(const bf16_t* __restrict__ Abf,
                                               const bf16_t* __restrict__ vbh,
                                               float* __restrict__ Av) {
    const int m    = blockIdx.x;
    const int wave = threadIdx.x >> 5;
    const int lane = threadIdx.x & 31;
    const int l16  = lane & 15;
    const int hiH  = lane >> 4;
    const int ksel = hiH * 8, kofb = hiH * 16;
    const bf16_t* A0row = Abf + ((size_t)(0 * MODES + m) * HID + wave * 16 + l16) * HID;
    const bf16_t* A1row = Abf + ((size_t)(1 * MODES + m) * HID + wave * 16 + l16) * HID;
    const bf16_t* vre   = vbh + ((size_t)(m * 2 + 0) * BATCH + l16) * HID;
    const bf16_t* vim   = vbh + ((size_t)(m * 2 + 1) * BATCH + l16) * HID;
    __builtin_prefetch(A0row, 0, 0);   // global_prefetch_b8
    __builtin_prefetch(A1row, 0, 0);
    v8f y0r = {}, y1i = {}, y1r = {}, y0i = {};
#pragma unroll
    for (int kt = 0; kt < 4; ++kt) {
        const int k0 = kt * 32;
        v16bf a0 = joinA(A0row, k0, ksel);
        v16bf a1 = joinA(A1row, k0, ksel);
        v16bf br = ld16(vre + k0 + kofb);
        v16bf bi = ld16(vim + k0 + kofb);
        y0r = wmma_bf16(a0, br, y0r);
        y1i = wmma_bf16(a1, bi, y1i);
        y1r = wmma_bf16(a1, br, y1r);
        y0i = wmma_bf16(a0, bi, y0i);
    }
    v8f avr = y0r - y1i;
    v8f avi = y1r - y0i;
    const int hbase = wave * 16 + hiH * 8;          // D rows = h_out, contiguous
    store8(Av + ((size_t)(m * 2 + 0) * BATCH + l16) * HID + hbase, avr);
    store8(Av + ((size_t)(m * 2 + 1) * BATCH + l16) * HID + hbase, avi);
}

// ------------------------------- band IDFT ---------------------------------
// z[row=(b,h)][x] = sum_m vre*Btre + vim*Btim. M=2048 rows, N=128, K=96.
// Writes output time-slice out[b][h][x][t] and z_bf[b][x][h] (h contiguous).
__global__ void __launch_bounds__(256) k_idft(const bf16_t* __restrict__ vbm,
                                              const bf16_t* __restrict__ Btre,
                                              const bf16_t* __restrict__ Btim,
                                              float* __restrict__ out,
                                              bf16_t* __restrict__ zbf, int tstep) {
    const int rowTile = blockIdx.x;            // 0..127
    const int colTile = threadIdx.x >> 5;      // 0..7
    const int lane = threadIdx.x & 31;
    const int l16 = lane & 15, hiH = lane >> 4;
    const int ksel = hiH * 8, kofb = hiH * 16;
    const int row  = rowTile * 16 + l16;
    const bf16_t* are = vbm + (size_t)(0 * ROWS + row) * MODES;
    const bf16_t* aim = vbm + (size_t)(1 * ROWS + row) * MODES;
    const int colx = colTile * 16 + l16;
    const bf16_t* bre = Btre + (size_t)colx * MODES;
    const bf16_t* bim = Btim + (size_t)colx * MODES;
    v8f acc = {};
#pragma unroll
    for (int kt = 0; kt < 3; ++kt) {
        const int k0 = kt * 32;
        acc = wmma_bf16(joinA(are, k0, ksel), ld16(bre + k0 + kofb), acc);
        acc = wmma_bf16(joinA(aim, k0, ksel), ld16(bim + k0 + kofb), acc);
    }
    const int rowBase = rowTile * 16 + hiH * 8;     // (b,h); tile never crosses b
    const int b = rowBase >> 7, h0 = rowBase & 127;
#pragma unroll
    for (int r = 0; r < 8; ++r)
        out[(size_t)((rowBase + r) * DX + colx) * DT + tstep] = acc[r];
    v8bf z8;
#pragma unroll
    for (int r = 0; r < 8; ++r) z8[r] = (bf16_t)acc[r];
    *reinterpret_cast<v8bf*>(zbf + ((size_t)(b * DX + colx) * HID + h0)) = z8;
}

// --------------------------- fused W_F/W_G GEMM ----------------------------
// P[b][o'][x] = sum_h Wcat[o'][h] * z[b][h][x].  M=640, N=128, K=128 per b.
__global__ void __launch_bounds__(256) k_fg(const bf16_t* __restrict__ Wcat,
                                            const bf16_t* __restrict__ zbf,
                                            float* __restrict__ P) {
    const int w = blockIdx.x * 8 + (threadIdx.x >> 5);   // 0..5119
    const int lane = threadIdx.x & 31;
    const int l16 = lane & 15, hiH = lane >> 4;
    const int ksel = hiH * 8, kofb = hiH * 16;
    const int b = w / 320, rem = w % 320;
    const int Mtile = rem >> 3, Ntile = rem & 7;
    const bf16_t* arow = Wcat + (size_t)(Mtile * 16 + l16) * HID;
    const int colx = Ntile * 16 + l16;
    const bf16_t* bcol = zbf + (size_t)(b * DX + colx) * HID;   // K=h contiguous
    v8f acc = {};
#pragma unroll
    for (int kt = 0; kt < 4; ++kt)
        acc = wmma_bf16(joinA(arow, kt * 32, ksel), ld16(bcol + kt * 32 + kofb), acc);
    const int obase = Mtile * 16 + hiH * 8;
#pragma unroll
    for (int r = 0; r < 8; ++r)
        P[(size_t)(b * 640 + obase + r) * DX + colx] = acc[r];
}

// ------------------- pointwise: H = tanh(F) + sum tanh(G)*dxi --------------
__global__ void k_h(const float* __restrict__ P, const float* __restrict__ dxi,
                    bf16_t* __restrict__ Hbf, int step) {
    int i = blockIdx.x * blockDim.x + threadIdx.x;
    if (i >= BATCH * HID * DX) return;
    int x = i & 127, o = (i >> 7) & 127, b = i >> 14;
    const float* Pb = P + (size_t)b * 640 * DX;
    float s = tanhf(Pb[o * DX + x]);
    const float* dx = dxi + (size_t)((step * BATCH + b) * DX + x) * NF;
#pragma unroll
    for (int n = 0; n < NF; ++n)
        s += tanhf(Pb[(HID + o * NF + n) * DX + x]) * dx[n];
    Hbf[i] = (bf16_t)s;                    // layout [b][h][x]
}

// -------------------- band DFT + Euler state update ------------------------
// hf[c][m][row] = sum_x H[row][x]*Wf_c[m][x]. M=2048 rows, N=96 modes, K=128.
// accumulate=1: v = v + Av + hf (Euler step); accumulate=0: v = hf (v0 init).
__global__ void __launch_bounds__(256) k_dft_update(const bf16_t* __restrict__ Hsrc,
                                                    const bf16_t* __restrict__ Wfre,
                                                    const bf16_t* __restrict__ Wfim,
                                                    const float* __restrict__ Av,
                                                    float* __restrict__ v,
                                                    int accumulate) {
    const int w = blockIdx.x * 8 + (threadIdx.x >> 5);   // 0..767
    const int lane = threadIdx.x & 31;
    const int l16 = lane & 15, hiH = lane >> 4;
    const int ksel = hiH * 8, kofb = hiH * 16;
    const int rowTile = w / 6, mTile = w % 6;
    const int row = rowTile * 16 + l16;
    const bf16_t* arow = Hsrc + (size_t)row * DX;        // K=x contiguous
    const int colm = mTile * 16 + l16;
    const bf16_t* bre = Wfre + (size_t)colm * DX;
    const bf16_t* bim = Wfim + (size_t)colm * DX;
    v8f ar = {}, ai = {};
#pragma unroll
    for (int kt = 0; kt < 4; ++kt) {
        v16bf a = joinA(arow, kt * 32, ksel);
        ar = wmma_bf16(a, ld16(bre + kt * 32 + kofb), ar);
        ai = wmma_bf16(a, ld16(bim + kt * 32 + kofb), ai);
    }
    const int rowBase = rowTile * 16 + hiH * 8;          // lane: fixed m, 8 rows
    float* pvr = v + (size_t)(colm * 2 + 0) * ROWS + rowBase;
    float* pvi = v + (size_t)(colm * 2 + 1) * ROWS + rowBase;
    if (accumulate) {
        const float* avr = Av + (size_t)(colm * 2 + 0) * ROWS + rowBase;
        const float* avi = Av + (size_t)(colm * 2 + 1) * ROWS + rowBase;
#pragma unroll
        for (int r = 0; r < 8; ++r) {
            ar[r] += pvr[r] + avr[r];
            ai[r] += pvi[r] + avi[r];
        }
    }
    store8(pvr, ar);
    store8(pvi, ai);
}

// ------------------------------- host side ---------------------------------

extern "C" void kernel_launch(void* const* d_in, const int* in_sizes, int n_in,
                              void* d_out, int out_size, void* d_ws, size_t ws_size,
                              hipStream_t stream) {
    const float* z0 = (const float*)d_in[0];
    const float* xi = (const float*)d_in[1];
    const float* A  = (const float*)d_in[2];
    const float* WF = (const float*)d_in[3];
    const float* WG = (const float*)d_in[4];
    float* out = (float*)d_out;
    char* ws = (char*)d_ws;

    // Workspace layout (bytes), ~20.2 MB total, resident in L2.
    size_t off = 0;
    auto take = [&](size_t bytes) { size_t o = off; off = (off + bytes + 255) & ~(size_t)255; return o; };
    bf16_t* Abf  = (bf16_t*)(ws + take((size_t)2 * MODES * HID * HID * 2));
    bf16_t* Wcat = (bf16_t*)(ws + take((size_t)640 * HID * 2));
    bf16_t* Wfre = (bf16_t*)(ws + take((size_t)MODES * DX * 2));
    bf16_t* Wfim = (bf16_t*)(ws + take((size_t)MODES * DX * 2));
    bf16_t* Btre = (bf16_t*)(ws + take((size_t)DX * MODES * 2));
    bf16_t* Btim = (bf16_t*)(ws + take((size_t)DX * MODES * 2));
    bf16_t* z0bf = (bf16_t*)(ws + take((size_t)BATCH * HID * DX * 2));
    float*  dxi  = (float*) (ws + take((size_t)NSTEP * BATCH * DX * NF * 4));
    float*  v    = (float*) (ws + take((size_t)MODES * 2 * ROWS * 4));
    float*  Avp  = (float*) (ws + take((size_t)MODES * 2 * ROWS * 4));
    bf16_t* vbh  = (bf16_t*)(ws + take((size_t)MODES * 2 * ROWS * 2));
    bf16_t* vbm  = (bf16_t*)(ws + take((size_t)2 * ROWS * MODES * 2));
    bf16_t* zbf  = (bf16_t*)(ws + take((size_t)BATCH * DX * HID * 2));
    bf16_t* Hbf  = (bf16_t*)(ws + take((size_t)BATCH * HID * DX * 2));
    float*  P    = (float*) (ws + take((size_t)BATCH * 640 * DX * 4));

    // ---- one-time (per launch) setup: bf16 operand prep + twiddles ----
    {
        int n = 2 * MODES * HID * HID;
        k_cvt<<<(n + 255) / 256, 256, 0, stream>>>(A, Abf, n);
    }
    k_wcat<<<(640 * HID + 255) / 256, 256, 0, stream>>>(WF, WG, Wcat);
    k_twiddle<<<(DX * MODES + 255) / 256, 256, 0, stream>>>(Wfre, Wfim, Btre, Btim);
    {
        int n = BATCH * HID * DX;
        k_cvt<<<(n + 255) / 256, 256, 0, stream>>>(z0, z0bf, n);
    }
    k_dxi<<<(NSTEP * BATCH * DX * NF + 255) / 256, 256, 0, stream>>>(xi, dxi);

    // v0 = band-limited shifted DFT of z0 (accumulate=0).
    k_dft_update<<<96, 256, 0, stream>>>(z0bf, Wfre, Wfim, Avp, v, 0);

    // ---- 63 Euler steps ----
    for (int step = 0; step < NSTEP; ++step) {
        k_cvtv <<<1536, 256, 0, stream>>>(v, vbh, vbm);
        k_modes<<<96,   256, 0, stream>>>(Abf, vbh, Avp);
        k_idft <<<128,  256, 0, stream>>>(vbm, Btre, Btim, out, zbf, step);
        k_fg   <<<640,  256, 0, stream>>>(Wcat, zbf, P);
        k_h    <<<1024, 256, 0, stream>>>(P, dxi, Hbf, step);
        k_dft_update<<<96, 256, 0, stream>>>(Hbf, Wfre, Wfim, Avp, v, 1);
    }

    // Final time slice t=63 from the last state.
    k_cvtv<<<1536, 256, 0, stream>>>(v, vbh, vbm);
    k_idft<<<128, 256, 0, stream>>>(vbm, Btre, Btim, out, zbf, NSTEP);
}
